// MambaBlock_76802605187530
// MI455X (gfx1250) — compile-verified, matrix operand order checked by reading, not run on hardware
//
#include <hip/hip_runtime.h>

// ---------------------------------------------------------------------------
// Mamba block for MI455X (gfx1250, wave32).
// GEMMs: bf16 V_WMMA_F32_16X16X32_BF16, f32 accumulate, LDS-staged tiles.
//        Block tile 128x64 (8 waves), wave tile 32x32 (4 accumulators,
//        a0/a1 x b0/b1 fragment reuse -> 4 WMMAs per 4 ds_load-pairs).
// Scan:  3-phase chunked parallel scan (chunk=128, 32 chunks per sequence).
// ---------------------------------------------------------------------------

#define D_MODEL 1024
#define D_STATE 128
#define D_CONV  4
#define DT_RANK 64
#define D_INNER 2048
#define BATCH   4
#define SEQ     4096
#define MTOT    (BATCH * SEQ)          // 16384 rows
#define CHUNK   128
#define NCHUNK  (SEQ / CHUNK)          // 32

typedef __attribute__((ext_vector_type(16))) __bf16 v16bf;
typedef __attribute__((ext_vector_type(8)))  __bf16 v8bf;
typedef __attribute__((ext_vector_type(8)))  float  v8f;

// float -> bf16, round-to-nearest-even
__device__ __forceinline__ __bf16 f2bf(float f) {
    unsigned u = __builtin_bit_cast(unsigned, f);
    u += 0x7FFFu + ((u >> 16) & 1u);
    unsigned short h = (unsigned short)(u >> 16);
    return __builtin_bit_cast(__bf16, h);
}

// Assemble a 16x32 (or 32x16) bf16 WMMA fragment for this lane from an LDS row.
// Elements 0..7  <-> K = 8*kh + j ; elements 8..15 <-> K = 16 + 8*kh + j.
__device__ __forceinline__ v16bf ldfrag(const unsigned short* __restrict__ row, int kh) {
    v8bf lo = *(const v8bf*)(row + kh * 8);
    v8bf hi = *(const v8bf*)(row + 16 + kh * 8);
    v16bf r;
#pragma unroll
    for (int i = 0; i < 8; ++i) { r[i] = lo[i]; r[i + 8] = hi[i]; }
    return r;
}

// ---------------------------------------------------------------------------
// GEMM: Out[M,N] = X[M,K] * W[N,K]^T   (M%128==0, N%64==0, K%32==0)
// Block: 256 threads = 8 waves; block tile 128x64.
// Wave w: wm=w&3 -> 32-row band, wn=w>>2 -> 32-col band. Wave tile 32x32:
// fragments a0,a1 (M halves) x b0,b1 (N halves) -> 4 WMMA accumulators.
// EPI==1 fuses: out = (acc + e_add*e_D[n]) * silu(e_z)   (the Mamba gate)
// ---------------------------------------------------------------------------
#define LDSPAD 56   // row stride in bf16 elems: 112B = 7*16B aligned, conflict-free

template <int EPI>
__global__ __launch_bounds__(256) void gemm_bf16_wmma(
    const float* __restrict__ X, const float* __restrict__ W,
    float* __restrict__ Out, int M, int N, int K,
    const float* __restrict__ e_add, const float* __restrict__ e_D,
    const float* __restrict__ e_z, int e_zstride)
{
    __shared__ __align__(16) unsigned short As[128 * LDSPAD];
    __shared__ __align__(16) unsigned short Bs[64 * LDSPAD];

    const int lane = threadIdx.x & 31;
    const int wave = threadIdx.x >> 5;
    const int wm   = wave & 3;        // 0..3 -> 32-row band
    const int wn   = wave >> 2;       // 0..1 -> 32-col band
    const int mblk = blockIdx.x * 128;
    const int nblk = blockIdx.y * 64;

    // cooperative LDS fill: A is 128x32 (two 8-elem chunks/thread), B is 64x32
    const int fr = threadIdx.x >> 2;          // 0..63
    const int fc = (threadIdx.x & 3) * 8;     // 0,8,16,24
    const float* pa0 = X + (size_t)(mblk + fr) * K + fc;        // rows 0..63
    const float* pa1 = X + (size_t)(mblk + 64 + fr) * K + fc;   // rows 64..127
    const float* pb  = W + (size_t)(nblk + fr) * K + fc;
    unsigned short* da0 = &As[fr * LDSPAD + fc];
    unsigned short* da1 = &As[(64 + fr) * LDSPAD + fc];
    unsigned short* db  = &Bs[fr * LDSPAD + fc];

    const int kh  = lane >> 4;                // K-half held by this lane
    const int t16 = lane & 15;                // row (A) / col (B) index
    const unsigned short* ra0 = &As[(wm * 32 + t16) * LDSPAD];
    const unsigned short* ra1 = &As[(wm * 32 + 16 + t16) * LDSPAD];
    const unsigned short* rb0 = &Bs[(wn * 32 + t16) * LDSPAD];
    const unsigned short* rb1 = &Bs[(wn * 32 + 16 + t16) * LDSPAD];

    v8f acc00 = {}, acc01 = {}, acc10 = {}, acc11 = {};

    for (int k0 = 0; k0 < K; k0 += 32) {
        {   // stage A and B tiles into LDS as bf16
            v8bf ta0, ta1, tb;
#pragma unroll
            for (int i = 0; i < 8; ++i) {
                ta0[i] = f2bf(pa0[i]);
                ta1[i] = f2bf(pa1[i]);
                tb[i]  = f2bf(pb[i]);
            }
            *(v8bf*)da0 = ta0;
            *(v8bf*)da1 = ta1;
            *(v8bf*)db  = tb;
            pa0 += 32; pa1 += 32; pb += 32;
        }
        __syncthreads();

        v16bf a0 = ldfrag(ra0, kh);
        v16bf a1 = ldfrag(ra1, kh);
        v16bf b0 = ldfrag(rb0, kh);
        v16bf b1 = ldfrag(rb1, kh);

        acc00 = __builtin_amdgcn_wmma_f32_16x16x32_bf16(
                    false, a0, false, b0, (short)0, acc00, false, false);
        acc01 = __builtin_amdgcn_wmma_f32_16x16x32_bf16(
                    false, a0, false, b1, (short)0, acc01, false, false);
        acc10 = __builtin_amdgcn_wmma_f32_16x16x32_bf16(
                    false, a1, false, b0, (short)0, acc10, false, false);
        acc11 = __builtin_amdgcn_wmma_f32_16x16x32_bf16(
                    false, a1, false, b1, (short)0, acc11, false, false);
        __syncthreads();
    }

    // D layout: VGPR v -> M = v + 8*kh, N = t16 (+ band offsets)
    const int nG0 = nblk + wn * 32 + t16;   // cols of b0 accs
    const int nG1 = nG0 + 16;               // cols of b1 accs
    const int mA  = mblk + wm * 32 + kh * 8;
    size_t r0 = (size_t)mA * N;             // row offset of acc0x[v=0]
    size_t r1 = r0 + (size_t)16 * N;        // row offset of acc1x[v=0]
    if (EPI == 0) {
#pragma unroll
        for (int v = 0; v < 8; ++v) {
            Out[r0 + nG0] = acc00[v];
            Out[r0 + nG1] = acc01[v];
            Out[r1 + nG0] = acc10[v];
            Out[r1 + nG1] = acc11[v];
            r0 += N; r1 += N;
        }
    } else {
        const float d0 = e_D[nG0], d1 = e_D[nG1];
        size_t z0 = (size_t)mA * e_zstride;
        size_t z1 = z0 + (size_t)16 * e_zstride;
#pragma unroll
        for (int v = 0; v < 8; ++v) {
            float y00 = acc00[v] + e_add[r0 + nG0] * d0;
            float y01 = acc01[v] + e_add[r0 + nG1] * d1;
            float y10 = acc10[v] + e_add[r1 + nG0] * d0;
            float y11 = acc11[v] + e_add[r1 + nG1] * d1;
            float g00 = e_z[z0 + nG0], g01 = e_z[z0 + nG1];
            float g10 = e_z[z1 + nG0], g11 = e_z[z1 + nG1];
            Out[r0 + nG0] = y00 * (g00 / (1.0f + __expf(-g00)));
            Out[r0 + nG1] = y01 * (g01 / (1.0f + __expf(-g01)));
            Out[r1 + nG0] = y10 * (g10 / (1.0f + __expf(-g10)));
            Out[r1 + nG1] = y11 * (g11 / (1.0f + __expf(-g11)));
            r0 += N; r1 += N; z0 += e_zstride; z1 += e_zstride;
        }
    }
}

// ---------------------------------------------------------------------------
// Depthwise causal conv1d over x_in = xz[:, :D_INNER] (row stride 2*D_INNER)
// ---------------------------------------------------------------------------
__global__ void k_conv(const float* __restrict__ xz, const float* __restrict__ w,
                       const float* __restrict__ bias, float* __restrict__ xc)
{
    size_t i = (size_t)blockIdx.x * blockDim.x + threadIdx.x; // over MTOT*D_INNER
    int d = (int)(i % D_INNER);
    size_t m = i / D_INNER;
    int l = (int)(m % SEQ);
    size_t b = m / SEQ;
    float acc = bias[d];
#pragma unroll
    for (int j = 0; j < D_CONV; ++j) {
        int ll = l - (D_CONV - 1) + j;
        if (ll >= 0)
            acc += w[d * D_CONV + j] * xz[(b * SEQ + ll) * (size_t)(2 * D_INNER) + d];
    }
    xc[m * D_INNER + d] = acc;
}

// ---------------------------------------------------------------------------
// delta = clip(softplus(dt_r @ dtW^T + dt_b), 1e-3, 0.1); A = -exp(A_log)
// A_bar = exp(delta*A); B_bar = (A_bar-1)/A * Bm
// One block (128 threads) per token row.
// ---------------------------------------------------------------------------
__global__ void k_ssm_prep(const float* __restrict__ ssm, const float* __restrict__ dtW,
                           const float* __restrict__ dtB, const float* __restrict__ A_log,
                           float* __restrict__ Abar, float* __restrict__ Bbar)
{
    __shared__ float dt[DT_RANK];
    const int row = blockIdx.x;
    const int n = threadIdx.x;
    if (n < DT_RANK) dt[n] = ssm[(size_t)row * (DT_RANK + 2 * D_STATE) + n];
    __syncthreads();

    float acc = dtB[n];
#pragma unroll 8
    for (int k = 0; k < DT_RANK; ++k) acc += dt[k] * dtW[n * DT_RANK + k];

    float delta = (acc > 20.0f) ? acc : log1pf(__expf(acc));
    delta = fminf(fmaxf(delta, 0.001f), 0.1f);

    float A  = -__expf(A_log[n]);
    float ab = __expf(delta * A);
    float As = (fabsf(A) > 1e-8f) ? A : 1.0f;
    float Bm = ssm[(size_t)row * (DT_RANK + 2 * D_STATE) + DT_RANK + n];
    float bb = (ab - 1.0f) / As * Bm;

    size_t o = (size_t)row * D_STATE + n;
    Abar[o] = ab;
    Bbar[o] = bb;
}

// ---------------------------------------------------------------------------
// Chunked scan, phase 1: local scan inside each chunk of 128 steps, also
// recording running products P (for carry fix-up) and chunk summaries.
// Block = (b, chunk), threads = state index n. Coalesced across n.
// ---------------------------------------------------------------------------
__global__ void k_scan1(const float* __restrict__ Abar, const float* __restrict__ Bbar,
                        float* __restrict__ hs, float* __restrict__ P,
                        float* __restrict__ Aprod, float* __restrict__ Hend)
{
    const int b = blockIdx.x / NCHUNK;
    const int c = blockIdx.x % NCHUNK;
    const int n = threadIdx.x;
    size_t base = ((size_t)b * SEQ + (size_t)c * CHUNK) * D_STATE + n;
    float h = 0.0f, p = 1.0f;
    for (int i = 0; i < CHUNK; ++i) {
        size_t o = base + (size_t)i * D_STATE;
        float a = Abar[o], bb = Bbar[o];
        h = fmaf(a, h, bb);
        p *= a;
        hs[o] = h;
        P[o]  = p;
    }
    size_t s = ((size_t)b * NCHUNK + c) * D_STATE + n;
    Aprod[s] = p;
    Hend[s]  = h;
}

// Phase 2: serial scan over the 32 chunk summaries -> carry-in per chunk.
__global__ void k_scan2(const float* __restrict__ Aprod, const float* __restrict__ Hend,
                        float* __restrict__ Carry)
{
    const int b = blockIdx.x;
    const int n = threadIdx.x;
    float carry = 0.0f;
    for (int c = 0; c < NCHUNK; ++c) {
        size_t s = ((size_t)b * NCHUNK + c) * D_STATE + n;
        Carry[s] = carry;
        carry = fmaf(Aprod[s], carry, Hend[s]);
    }
}

// Phase 3: h_true = h_local + P * carry_in(chunk)
__global__ void k_scan3(float* __restrict__ hs, const float* __restrict__ P,
                        const float* __restrict__ Carry)
{
    size_t i = (size_t)blockIdx.x * blockDim.x + threadIdx.x; // MTOT*D_STATE
    int n = (int)(i % D_STATE);
    size_t m = i / D_STATE;
    int l = (int)(m % SEQ);
    size_t b = m / SEQ;
    int c = l / CHUNK;
    hs[i] = fmaf(P[i], Carry[((size_t)b * NCHUNK + c) * D_STATE + n], hs[i]);
}

// ---------------------------------------------------------------------------
extern "C" void kernel_launch(void* const* d_in, const int* in_sizes, int n_in,
                              void* d_out, int out_size, void* d_ws, size_t ws_size,
                              hipStream_t stream)
{
    const float* x      = (const float*)d_in[0];
    const float* w_in   = (const float*)d_in[1];   // (4096, 1024)
    const float* conv_w = (const float*)d_in[2];   // (2048, 1, 4)
    const float* conv_b = (const float*)d_in[3];   // (2048,)
    const float* w_xp   = (const float*)d_in[4];   // (320, 2048)
    const float* w_dt   = (const float*)d_in[5];   // (128, 64)
    const float* b_dt   = (const float*)d_in[6];   // (128,)
    const float* A_log  = (const float*)d_in[7];   // (128,)
    const float* Dvec   = (const float*)d_in[8];   // (2048,)
    const float* w_C    = (const float*)d_in[9];   // (2048, 128)
    const float* w_out  = (const float*)d_in[10];  // (1024, 2048)
    float* out = (float*)d_out;

    // workspace carve-up (floats)
    float* ws    = (float*)d_ws;
    float* xz    = ws;                                   // 16384*4096
    float* xc    = xz    + (size_t)MTOT * 2 * D_INNER;   // 16384*2048 (reused as gated y)
    float* ssm   = xc    + (size_t)MTOT * D_INNER;       // 16384*320
    float* Abar  = ssm   + (size_t)MTOT * (DT_RANK + 2 * D_STATE);
    float* Bbar  = Abar  + (size_t)MTOT * D_STATE;
    float* hs    = Bbar  + (size_t)MTOT * D_STATE;
    float* Pp    = hs    + (size_t)MTOT * D_STATE;
    float* Aprod = Pp    + (size_t)MTOT * D_STATE;
    float* Hend  = Aprod + (size_t)BATCH * NCHUNK * D_STATE;
    float* Carry = Hend  + (size_t)BATCH * NCHUNK * D_STATE;

    // 1) in_proj: xz = x @ w_in^T                      (16384 x 4096, K=1024)
    gemm_bf16_wmma<0><<<dim3(MTOT / 128, (2 * D_INNER) / 64), 256, 0, stream>>>(
        x, w_in, xz, MTOT, 2 * D_INNER, D_MODEL, nullptr, nullptr, nullptr, 0);

    // 2) depthwise causal conv on x_in part of xz -> xc
    k_conv<<<(unsigned)(((size_t)MTOT * D_INNER) / 256), 256, 0, stream>>>(
        xz, conv_w, conv_b, xc);

    // 3) x_proj: ssm = xc @ w_xp^T                     (16384 x 320, K=2048)
    gemm_bf16_wmma<0><<<dim3(MTOT / 128, (DT_RANK + 2 * D_STATE) / 64), 256, 0, stream>>>(
        xc, w_xp, ssm, MTOT, DT_RANK + 2 * D_STATE, D_INNER, nullptr, nullptr, nullptr, 0);

    // 4) delta/softplus/clip, A_bar, B_bar
    k_ssm_prep<<<MTOT, D_STATE, 0, stream>>>(ssm, w_dt, b_dt, A_log, Abar, Bbar);

    // 5) chunked parallel scan: h_t = A_bar_t * h_{t-1} + B_bar_t
    k_scan1<<<BATCH * NCHUNK, D_STATE, 0, stream>>>(Abar, Bbar, hs, Pp, Aprod, Hend);
    k_scan2<<<BATCH, D_STATE, 0, stream>>>(Aprod, Hend, Carry);
    k_scan3<<<(unsigned)(((size_t)MTOT * D_STATE) / 256), 256, 0, stream>>>(hs, Pp, Carry);

    // 6) y = hs @ C^T, fused: (y + xc*D) * silu(z)  -> xc   (16384 x 2048, K=128)
    gemm_bf16_wmma<1><<<dim3(MTOT / 128, D_INNER / 64), 256, 0, stream>>>(
        hs, w_C, xc, MTOT, D_INNER, D_STATE,
        /*e_add=*/xc, /*e_D=*/Dvec, /*e_z=*/xz + D_INNER, /*e_zstride=*/2 * D_INNER);

    // 7) out_proj: out = g @ w_out^T                   (16384 x 1024, K=2048)
    gemm_bf16_wmma<0><<<dim3(MTOT / 128, D_MODEL / 64), 256, 0, stream>>>(
        xc, w_out, out, MTOT, D_MODEL, D_INNER, nullptr, nullptr, nullptr, 0);
}
